// AttachmentPredictor_56470230007954
// MI455X (gfx1250) — matrix operand order repaired and checked by reading, not compiled, not run on hardware
//
#include <hip/hip_runtime.h>
#include <hip/hip_bf16.h>

// AMD MI455X / gfx1250 — fused attachment-predictor pipeline.
// GEMMs run on V_WMMA_F32_16X16X32_F16 (f16 in, f32 accumulate).
// B operands are pre-packed so each lane's fragment is one 32-byte load
// addressed as [uniform SGPR base] + [32-bit lane offset] + [imm] (GVS form),
// keeping the WMMA inner loop free of 64-bit VALU address math (and the
// WMMA->VALU hazard v_nops it forces).

typedef __attribute__((ext_vector_type(16))) _Float16 v16h;
typedef __attribute__((ext_vector_type(8)))  float    v8f;

#define B_    64
#define T_    2048
#define TH_   2046          // T-2 head positions
#define D_    1024
#define P_    512
#define BM    64            // rows of head positions per block
#define SAP   136           // sA row pitch in halfs (128 + 8 pad -> 4-bank rotation/row)
#define SCP   520           // sC row pitch in halfs (512 + 8 pad)

__device__ __forceinline__ float fast_tanh(float x) {
  x = fminf(fmaxf(x, -15.f), 15.f);         // avoid inf/inf
  float t = __expf(2.f * x);                // v_exp_f32
  return (t - 1.f) / (t + 1.f);
}

// -------------------------------------------------------------------------
// Pack a row-major f32 weight W[K][N] into WMMA-B fragment order (f16):
// for tile (kt,nt) and lane L: 16 contiguous halfs e=0..15 hold
//   W[kt*32 + e + 16*(L>=16)][nt*16 + (L&15)]
// so the fused kernel fetches each B fragment as one 32-byte load.
// -------------------------------------------------------------------------
__global__ void pack_w(const float* __restrict__ W, _Float16* __restrict__ Pk,
                       int K, int N) {
  const int gid  = blockIdx.x * blockDim.x + threadIdx.x;
  const int lane = gid & 31;
  const int tile = gid >> 5;
  const int ntn  = N >> 4;
  if (tile >= (K >> 5) * ntn) return;
  const int nt = tile % ntn;
  const int kt = tile / ntn;
  const int n  = nt * 16 + (lane & 15);
  const int kb = kt * 32 + ((lane >> 4) << 4);
  _Float16* dst = Pk + ((size_t)tile * 32 + lane) * 16;
  #pragma unroll
  for (int e = 0; e < 16; e++) dst[e] = (_Float16)W[(size_t)(kb + e) * N + n];
}

// -------------------------------------------------------------------------
// bias[b][p] = prep_enc[b]·proj_prep[:,p] + child_enc[b]·proj_child[:,p]
// -------------------------------------------------------------------------
__global__ __launch_bounds__(256) void bias_kernel(
    const float* __restrict__ x, const float* __restrict__ Wp,
    const float* __restrict__ Wc, float* __restrict__ bias) {
  __shared__ float sx[2 * D_];
  const int b = blockIdx.x, tid = threadIdx.x;
  const float* xp = x + ((size_t)b * T_ + (T_ - 2)) * D_;  // prep row then child row
  for (int i = tid; i < 2 * D_; i += 256) sx[i] = xp[i];
  __syncthreads();
  for (int p = tid; p < P_; p += 256) {
    float acc = 0.f;
    for (int d = 0; d < D_; d++)
      acc += sx[d] * Wp[(size_t)d * P_ + p] + sx[D_ + d] * Wc[(size_t)d * P_ + p];
    bias[b * P_ + p] = acc;
  }
}

// -------------------------------------------------------------------------
// Fused: composed = tanh(head@proj_head + bias); x2 tanh(.@W); score; exp*mask
// Block: 256 threads = 8 waves. Wave (ni = w&3, mg = w>>2) owns
// rows [mg*32, mg*32+32) x cols [ni*128, ni*128+128) as 2x8 WMMA tiles.
// -------------------------------------------------------------------------
__global__ __launch_bounds__(256) void fused_kernel(
    const float* __restrict__ x,
    const _Float16* __restrict__ pkA,   // packed proj_head  (K=1024,N=512)
    const _Float16* __restrict__ pkW0,  // packed hidden_W[0] (512x512)
    const _Float16* __restrict__ pkW1,  // packed hidden_W[1] (512x512)
    const float* __restrict__ bias,     // [64][512]
    const float* __restrict__ scorer,   // [512]
    const int* __restrict__ mask,       // [64][2048]
    float* __restrict__ out)            // [64][2046] masked exp (pre-normalize)
{
  __shared__ _Float16 sA[BM * SAP];     // x chunk, f16   (~17.4 KB)
  __shared__ _Float16 sC[BM * SCP];     // composed, f16  (~66.6 KB) — CDNA5 WGP: 320KB LDS

  const int b    = blockIdx.y;
  const int t0   = blockIdx.x * BM;
  const int tid  = threadIdx.x;
  const int lane = tid & 31;
  const int w    = tid >> 5;
  const int ni   = w & 3;
  const int mg   = w >> 2;
  const int lm   = lane & 15;
  const int hh   = lane >> 4;

  // 32-bit divergent components (uniform 64-bit bases stay in SGPRs -> GVS loads)
  const unsigned int bo   = (unsigned int)(ni * 4096 + lane * 16);        // B frag, halfs
  const unsigned int xoff = (unsigned int)((tid >> 2) * D_ + (tid & 3) * 32); // x stage, floats

  v8f acc[2][8];
  #pragma unroll
  for (int s = 0; s < 2; s++)
    #pragma unroll
    for (int i = 0; i < 8; i++)
      #pragma unroll
      for (int r = 0; r < 8; r++) acc[s][i][r] = 0.f;

  const float* xbase = x + (size_t)((size_t)b * T_ + t0) * D_;  // uniform

  // ---------------- Stage 1: (64x1024) @ (1024x512), K chunks of 128 ------
  for (int kc = 0; kc < 8; kc++) {
    {   // cooperative f32->f16 stage of the x chunk (64 rows x 128 cols)
      const float* src = xbase + xoff + kc * 128;
      _Float16* dst = &sA[(tid >> 2) * SAP + (tid & 3) * 32];
      #pragma unroll
      for (int u = 0; u < 32; u++) dst[u] = (_Float16)src[u];
      if (kc < 7) __builtin_prefetch(src + 128, 0, 3);   // pull next chunk toward WGP
    }
    __syncthreads();
    const _Float16* pb1 = pkA + (size_t)kc * 65536;      // uniform, advances via SALU
    #pragma unroll
    for (int ks = 0; ks < 4; ks++) {
      union { v16h v; unsigned int u[8]; } Af[2];
      #pragma unroll
      for (int s = 0; s < 2; s++) {
        const int arow = mg * 32 + s * 16 + lm;
        #pragma unroll
        for (int j = 0; j < 8; j++) {   // 16-bit A pairs, contiguous k = 16*hh + 2j (same order as pack_w)
          const int k0 = ks * 32 + hh * 16 + j * 2;
          Af[s].u[j] = *(const unsigned int*)&sA[arow * SAP + k0];
        }
      }
      #pragma unroll
      for (int nt = 0; nt < 8; nt++) {
        // halfs: ks*16384 + nt*512 are compile-time immediates; bo is the lane offset
        const v16h Bf = *(const v16h*)(pb1 + bo + (ks * 16384 + nt * 512));
        acc[0][nt] = __builtin_amdgcn_wmma_f32_16x16x32_f16(
            false, Af[0].v, false, Bf, (short)0, acc[0][nt], false, false);
        acc[1][nt] = __builtin_amdgcn_wmma_f32_16x16x32_f16(
            false, Af[1].v, false, Bf, (short)0, acc[1][nt], false, false);
      }
    }
    __syncthreads();
  }
  // epilogue 1: +bias, tanh, write composed to sC as f16
  #pragma unroll
  for (int nt = 0; nt < 8; nt++) {
    const int n  = ni * 128 + nt * 16 + lm;
    const float bv = bias[b * P_ + n];
    #pragma unroll
    for (int s = 0; s < 2; s++)
      #pragma unroll
      for (int r = 0; r < 8; r++) {
        const int m = mg * 32 + s * 16 + r + hh * 8;     // C layout: VGPR r, lane-half
        sC[m * SCP + n] = (_Float16)fast_tanh(acc[s][nt][r] + bv);
      }
  }
  __syncthreads();

  // ---------------- Stages 2 & 3: (64x512) @ (512x512), tanh ---------------
  const _Float16* pkW[2] = { pkW0, pkW1 };
  for (int layer = 0; layer < 2; layer++) {
    #pragma unroll
    for (int s = 0; s < 2; s++)
      #pragma unroll
      for (int i = 0; i < 8; i++)
        #pragma unroll
        for (int r = 0; r < 8; r++) acc[s][i][r] = 0.f;
    const _Float16* pkWl = pkW[layer];
    for (int ks = 0; ks < 16; ks++) {
      const _Float16* pb = pkWl + (size_t)ks * 16384;    // uniform SALU increment
      union { v16h v; unsigned int u[8]; } Af[2];
      #pragma unroll
      for (int s = 0; s < 2; s++) {
        const int arow = mg * 32 + s * 16 + lm;
        #pragma unroll
        for (int j = 0; j < 8; j++) {
          const int k0 = ks * 32 + hh * 16 + j * 2;
          Af[s].u[j] = *(const unsigned int*)&sC[arow * SCP + k0];
        }
      }
      #pragma unroll
      for (int nt = 0; nt < 8; nt++) {
        const v16h Bf = *(const v16h*)(pb + bo + nt * 512);
        acc[0][nt] = __builtin_amdgcn_wmma_f32_16x16x32_f16(
            false, Af[0].v, false, Bf, (short)0, acc[0][nt], false, false);
        acc[1][nt] = __builtin_amdgcn_wmma_f32_16x16x32_f16(
            false, Af[1].v, false, Bf, (short)0, acc[1][nt], false, false);
      }
    }
    __syncthreads();                      // all reads of sC done before overwrite
    #pragma unroll
    for (int nt = 0; nt < 8; nt++) {
      const int n = ni * 128 + nt * 16 + lm;
      #pragma unroll
      for (int s = 0; s < 2; s++)
        #pragma unroll
        for (int r = 0; r < 8; r++) {
          const int m = mg * 32 + s * 16 + r + hh * 8;
          sC[m * SCP + n] = (_Float16)fast_tanh(acc[s][nt][r]);
        }
    }
    __syncthreads();
  }

  // ---------------- Stage 4: scores -> masked exp --------------------------
  {   // wave w handles rows w*8..w*8+7; 4 lanes per row, shfl reduce
    const int rowl = w * 8 + (lane >> 2);
    const int seg  = lane & 3;
    float s = 0.f;
    const _Float16* cr = &sC[rowl * SCP + seg * 128];
    const float* sc = scorer + seg * 128;
    #pragma unroll 8
    for (int k = 0; k < 128; k++) s += (float)cr[k] * sc[k];
    s += __shfl_xor(s, 1, 32);
    s += __shfl_xor(s, 2, 32);
    if (seg == 0) {
      const int t = t0 + rowl;
      if (t < TH_) {
        const int mv = mask[b * T_ + t];
        out[(size_t)b * TH_ + t] = mv ? __expf(s) : 0.f;
      }
    }
  }
}

// -------------------------------------------------------------------------
// Per-batch softmax normalization: out /= (sum over T + EPS)
// -------------------------------------------------------------------------
__global__ __launch_bounds__(256) void normalize_kernel(float* __restrict__ out) {
  __shared__ float sred[256];
  const int b = blockIdx.x, tid = threadIdx.x;
  float* o = out + (size_t)b * TH_;
  float s = 0.f;
  for (int i = tid; i < TH_; i += 256) s += o[i];
  sred[tid] = s;
  __syncthreads();
  for (int k = 128; k > 0; k >>= 1) {
    if (tid < k) sred[tid] += sred[tid + k];
    __syncthreads();
  }
  const float inv = 1.f / (sred[0] + 1e-7f);
  for (int i = tid; i < TH_; i += 256) o[i] *= inv;
}

// -------------------------------------------------------------------------
extern "C" void kernel_launch(void* const* d_in, const int* in_sizes, int n_in,
                              void* d_out, int out_size, void* d_ws, size_t ws_size,
                              hipStream_t stream) {
  (void)in_sizes; (void)n_in; (void)out_size; (void)ws_size;
  const float* x          = (const float*)d_in[0];
  const float* proj_head  = (const float*)d_in[1];
  const float* proj_prep  = (const float*)d_in[2];
  const float* proj_child = (const float*)d_in[3];
  const float* hidden_W   = (const float*)d_in[4];
  const float* scorer     = (const float*)d_in[5];
  const int*   mask       = (const int*)d_in[6];
  float* out = (float*)d_out;

  // workspace: bias(128KB) | pkA(1MB f16) | pkW0(512KB) | pkW1(512KB) ~= 2.2MB
  char* ws = (char*)d_ws;
  float*    bias = (float*)ws;
  _Float16* pkA  = (_Float16*)(ws + (size_t)B_ * P_ * sizeof(float));
  _Float16* pkW0 = pkA  + (size_t)D_ * P_;
  _Float16* pkW1 = pkW0 + (size_t)P_ * P_;

  // 1) pack weights into WMMA B-fragment order (f16)
  pack_w<<<((D_/32)*(P_/16)*32) / 256, 256, 0, stream>>>(proj_head, pkA, D_, P_);
  pack_w<<<((P_/32)*(P_/16)*32) / 256, 256, 0, stream>>>(hidden_W,            pkW0, P_, P_);
  pack_w<<<((P_/32)*(P_/16)*32) / 256, 256, 0, stream>>>(hidden_W + P_ * P_,  pkW1, P_, P_);

  // 2) column bias from prep/child rows
  bias_kernel<<<B_, 256, 0, stream>>>(x, proj_prep, proj_child, bias);

  // 3) fused WMMA pipeline -> masked exp(scores)
  dim3 grid((TH_ + BM - 1) / BM, B_);   // 32 x 64 blocks
  fused_kernel<<<grid, 256, 0, stream>>>(x, pkA, pkW0, pkW1, bias, scorer, mask, out);

  // 4) per-batch normalization
  normalize_kernel<<<B_, 256, 0, stream>>>(out);
}
